// SynthFirstBlock_85169201480024
// MI455X (gfx1250) — compile-verified, matrix operand order checked by reading, not run on hardware
//
#include <hip/hip_runtime.h>

typedef __attribute__((ext_vector_type(16))) _Float16 v16h;
typedef __attribute__((ext_vector_type(8)))  _Float16 v8h;
typedef __attribute__((ext_vector_type(8)))  float    v8f;
typedef __attribute__((ext_vector_type(4)))  float    v4f;

#define LREL(x) ((x) >= 0.0f ? (x) : 0.2f * (x))
#define SPW 8            // samples per workgroup in the conv kernel
#define LDS_STRIDE 520   // 512 + 8 halves pad -> bank-conflict-free B loads

static __device__ __forceinline__ v16h pack16(v8h lo, v8h hi) {
  v16h r;
#pragma unroll
  for (int i = 0; i < 8; ++i) { r[i] = lo[i]; r[i + 8] = hi[i]; }
  return r;
}

static __device__ __forceinline__ v16h cvt16(v4f a0, v4f a1, v4f a2, v4f a3) {
  v16h r;
#pragma unroll
  for (int i = 0; i < 4; ++i) {
    r[i]      = (_Float16)a0[i];
    r[i + 4]  = (_Float16)a1[i];
    r[i + 8]  = (_Float16)a2[i];
    r[i + 12] = (_Float16)a3[i];
  }
  return r;
}

// ---- prep: conv_w (512,512,3,3) f32 -> cwh[t][co][ci] f16, scaled by 1/48 ----
__global__ __launch_bounds__(256) void prep_conv(const float* __restrict__ cw,
                                                 _Float16* __restrict__ cwh) {
  int idx = blockIdx.x * 256 + threadIdx.x;
  if (idx >= 9 * 512 * 512) return;
  int t   = idx >> 18;        // / (512*512)
  int rem = idx & 262143;
  int co  = rem >> 9;
  int ci  = rem & 511;
  cwh[idx] = (_Float16)(cw[(co * 512 + ci) * 9 + t] * (1.0f / 48.0f));
}

// ---- prep: a{1,2}_w (512,1024) f32 -> aT{1,2}[n][k] f16 (transposed) ----
__global__ __launch_bounds__(256) void prep_at(const float* __restrict__ a1,
                                               const float* __restrict__ a2,
                                               _Float16* __restrict__ aT1,
                                               _Float16* __restrict__ aT2) {
  int idx = blockIdx.x * 256 + threadIdx.x;
  if (idx >= 2 * 1024 * 512) return;
  int sel = idx >> 19;
  int rem = idx & 524287;
  int n = rem >> 9;
  int k = rem & 511;
  const float* src = sel ? a2 : a1;
  _Float16*    dst = sel ? aT2 : aT1;
  dst[rem] = (_Float16)src[k * 1024 + n];
}

// ---- style = w @ a_w + a_b  via WMMA; one wave per 16x16 tile, K=512 ----
__global__ __launch_bounds__(32) void style_gemm(
    const float* __restrict__ w1, const float* __restrict__ w2,
    const _Float16* __restrict__ aT1, const _Float16* __restrict__ aT2,
    const float* __restrict__ ab1, const float* __restrict__ ab2,
    float* __restrict__ st1, float* __restrict__ st2) {
  const float* w;  const _Float16* aT; const float* ab; float* st;
  if (blockIdx.z == 0) { w = w1; aT = aT1; ab = ab1; st = st1; }
  else                 { w = w2; aT = aT2; ab = ab2; st = st2; }

  const int lane = threadIdx.x;
  const int lm = lane & 15, h = lane >> 4;
  const int m0 = blockIdx.x * 16;
  const int n0 = blockIdx.y * 16;

  const float*    arow = w  + (size_t)(m0 + lm) * 512;
  const _Float16* brow = aT + (size_t)(n0 + lm) * 512;

  v8f acc = (v8f){0.f, 0.f, 0.f, 0.f, 0.f, 0.f, 0.f, 0.f};
  for (int kb = 0; kb < 512; kb += 32) {
    const int ka = kb + h * 8;
    v4f a0 = *(const v4f*)(arow + ka);
    v4f a1 = *(const v4f*)(arow + ka + 4);
    v4f a2 = *(const v4f*)(arow + ka + 16);
    v4f a3 = *(const v4f*)(arow + ka + 20);
    v16h A = cvt16(a0, a1, a2, a3);
    v16h B = pack16(*(const v8h*)(brow + ka), *(const v8h*)(brow + ka + 16));
    acc = __builtin_amdgcn_wmma_f32_16x16x32_f16(false, A, false, B, (short)0,
                                                 acc, false, false);
  }
  const float bias = ab[n0 + lm];
#pragma unroll
  for (int r = 0; r < 8; ++r)
    st[(size_t)(m0 + r + 8 * h) * 1024 + n0 + lm] = acc[r] + bias;
}

// ---- fused: AdaIN1 -> LDS(f16) -> conv3x3 via WMMA -> AdaIN2 -> out ----
__global__ __launch_bounds__(256) void synth_main(
    const float* __restrict__ noise1, const float* __restrict__ noise2,
    const float* __restrict__ base,   const float* __restrict__ conv_b,
    const float* __restrict__ ns1,    const float* __restrict__ ns2,
    const float* __restrict__ style1, const float* __restrict__ style2,
    const _Float16* __restrict__ cwh, float* __restrict__ out) {
  extern __shared__ char smem_raw[];
  _Float16* xlds = (_Float16*)smem_raw;  // [SPW][16][LDS_STRIDE]

  const int tid = threadIdx.x;
  const int b0  = blockIdx.x * SPW;

  // Phase 1: x1 = AdaIN1(leaky(base + ns1*noise1)), store f16 transposed [s][p][c]
  {
    const int s  = tid >> 5;          // sample within WG
    const int c0 = (tid & 31) * 16;   // 16 channels per thread
    const int b  = b0 + s;
    float nz[16];
#pragma unroll
    for (int p = 0; p < 16; ++p) nz[p] = noise1[b * 16 + p];
#pragma unroll 1
    for (int i = 0; i < 16; ++i) {
      const int c = c0 + i;
      const float nsc = ns1[c];
      float v[16];
      float mean = 0.f;
#pragma unroll
      for (int p = 0; p < 16; ++p) {
        float xv = base[c * 16 + p] + nsc * nz[p];
        xv = LREL(xv);
        v[p] = xv;
        mean += xv;
      }
      mean *= 0.0625f;
      float var = 0.f;
#pragma unroll
      for (int p = 0; p < 16; ++p) { float d = v[p] - mean; var += d * d; }
      var *= 0.0625f;
      const float ys = style1[b * 1024 + c];
      const float yb = style1[b * 1024 + 512 + c];
      const float sc = ys * rsqrtf(var + 1e-5f);
      _Float16* dst = xlds + (s * 16) * LDS_STRIDE + c;
#pragma unroll
      for (int p = 0; p < 16; ++p)
        dst[p * LDS_STRIDE] = (_Float16)((v[p] - mean) * sc + yb);
    }
  }
  __syncthreads();

  // Phase 2: conv as sum of 9 shifted GEMMs, WMMA 16x16x32 f16, f32 acc
  const int lane = tid & 31;
  const int wid  = tid >> 5;
  const int lm   = lane & 15;
  const int h    = lane >> 4;
  const int py   = lm >> 2, px = lm & 3;

  for (int cb = 0; cb < 4; ++cb) {
    const int co0 = (cb * 8 + wid) * 16;
    v8f acc[SPW];
#pragma unroll
    for (int s = 0; s < SPW; ++s)
      acc[s] = (v8f){0.f, 0.f, 0.f, 0.f, 0.f, 0.f, 0.f, 0.f};

    for (int t = 0; t < 9; ++t) {
      const int sy = py + (t / 3) - 1;
      const int sx = px + (t % 3) - 1;
      const bool valid = ((unsigned)sy < 4u) && ((unsigned)sx < 4u);
      const int  sp = valid ? (sy * 4 + sx) : 0;
      const _Float16* arow = cwh + (size_t)(t * 512 + co0 + lm) * 512;
      for (int kb = 0; kb < 512; kb += 32) {
        const int ka = kb + h * 8;
        v16h A = pack16(*(const v8h*)(arow + ka), *(const v8h*)(arow + ka + 16));
#pragma unroll
        for (int s = 0; s < SPW; ++s) {
          v16h B = {};
          if (valid) {
            const _Float16* bp = xlds + (s * 16 + sp) * LDS_STRIDE + ka;
            B = pack16(*(const v8h*)bp, *(const v8h*)(bp + 16));
          }
          acc[s] = __builtin_amdgcn_wmma_f32_16x16x32_f16(
              false, A, false, B, (short)0, acc[s], false, false);
        }
      }
    }

    // Epilogue: +bias +noise2, leaky, AdaIN2, store
#pragma unroll 1
    for (int s = 0; s < SPW; ++s) {
      const int b = b0 + s;
      const float n2 = noise2[b * 16 + lm];
#pragma unroll
      for (int r = 0; r < 8; ++r) {
        const int co = co0 + r + 8 * h;
        float val = acc[s][r] + conv_b[co] + ns2[co] * n2;
        val = LREL(val);
        float sum = val, sq = val * val;
#pragma unroll
        for (int m = 1; m < 16; m <<= 1) {
          sum += __shfl_xor(sum, m, 32);
          sq  += __shfl_xor(sq,  m, 32);
        }
        const float mean = sum * 0.0625f;
        const float var  = sq * 0.0625f - mean * mean;
        const float ys = style2[b * 1024 + co];
        const float yb = style2[b * 1024 + 512 + co];
        out[((size_t)b * 512 + co) * 16 + lm] =
            ys * ((val - mean) * rsqrtf(var + 1e-5f)) + yb;
      }
    }
  }
}

extern "C" void kernel_launch(void* const* d_in, const int* in_sizes, int n_in,
                              void* d_out, int out_size, void* d_ws, size_t ws_size,
                              hipStream_t stream) {
  (void)in_sizes; (void)n_in; (void)out_size; (void)ws_size;
  const float* w1     = (const float*)d_in[0];
  const float* w2     = (const float*)d_in[1];
  const float* noise1 = (const float*)d_in[2];
  const float* noise2 = (const float*)d_in[3];
  const float* base   = (const float*)d_in[4];
  const float* conv_w = (const float*)d_in[5];
  const float* conv_b = (const float*)d_in[6];
  const float* ns1    = (const float*)d_in[7];
  const float* ns2    = (const float*)d_in[8];
  const float* a1_w   = (const float*)d_in[9];
  const float* a1_b   = (const float*)d_in[10];
  const float* a2_w   = (const float*)d_in[11];
  const float* a2_b   = (const float*)d_in[12];
  float* out = (float*)d_out;

  char* ws = (char*)d_ws;
  float*    style1 = (float*)(ws);                          // 16,777,216 B
  float*    style2 = (float*)(ws + 16777216);               // 16,777,216 B
  _Float16* cwh    = (_Float16*)(ws + 33554432);            //  4,718,592 B
  _Float16* aT1    = (_Float16*)(ws + 38273024);            //  1,048,576 B
  _Float16* aT2    = (_Float16*)(ws + 39321600);            //  1,048,576 B

  prep_conv<<<9216, 256, 0, stream>>>(conv_w, cwh);
  prep_at<<<4096, 256, 0, stream>>>(a1_w, a2_w, aT1, aT2);

  dim3 g(4096 / 16, 1024 / 16, 2);
  style_gemm<<<g, 32, 0, stream>>>(w1, w2, aT1, aT2, a1_b, a2_b, style1, style2);

  const size_t lds_bytes = (size_t)SPW * 16 * LDS_STRIDE * sizeof(_Float16); // 133,120
  synth_main<<<4096 / SPW, 256, lds_bytes, stream>>>(
      noise1, noise2, base, conv_b, ns1, ns2, style1, style2, cwh, out);
}
